// DHMSA_27522150433348
// MI455X (gfx1250) — matrix operand order repaired and checked by reading, not compiled, not run on hardware
//
#include <hip/hip_runtime.h>
#include <math.h>

// ---------------------------------------------------------------------------
// Halo-window attention (Swin-v2 style) for MI455X / gfx1250, wave32 + WMMA.
// B=2, H=W=188, C=256, HEADS=8, HD=32, W_CUR=8, HALO=16, PH=PW=192, NW=576.
// ---------------------------------------------------------------------------

typedef __attribute__((ext_vector_type(16))) _Float16 v16h;
typedef __attribute__((ext_vector_type(8)))  _Float16 v8h;
typedef __attribute__((ext_vector_type(8)))  float    v8f;

#define BQ 2
#define HQ 188
#define CQ 256
#define PHQ 192
#define NWIN 24              // windows per side
#define TOK (BQ*HQ*HQ)       // 70688 tokens
#define TOKPB (HQ*HQ)        // 35344

union FragU { v16h v; v8h h[2]; };

__device__ __forceinline__ v8f wmma_f16(const FragU& a, const FragU& b, v8f c) {
  return __builtin_amdgcn_wmma_f32_16x16x32_f16(false, a.v, false, b.v, (short)0, c, false, false);
}

__device__ __forceinline__ unsigned pack2h(float a, float b) {
  union { _Float16 h[2]; unsigned u; } t;
  t.h[0] = (_Float16)a; t.h[1] = (_Float16)b;
  return t.u;
}

// ---------------------------------------------------------------------------
// Generic GEMM: C[M x N] = A[M x 256] * B[256 x N], f32 in/out, f16 WMMA math.
// Block = 256 threads (8 waves). Block tile: 16(M) x 128(N). K step = 32.
// Double-buffered LDS; next tile is loaded to registers while WMMA runs.
// MODE 0: A rows dense, stride 256 (QKV gemm, A = x).
// MODE 1: A rows gathered with 188->192 crop mapping (proj gemm, A = attn_out).
// ---------------------------------------------------------------------------
template <int N, int MODE>
__global__ __launch_bounds__(256) void gemm16x128_kernel(
    const float* __restrict__ A, const float* __restrict__ B,
    float* __restrict__ C)
{
  __shared__ alignas(16) _Float16 AhL[2][16 * 40];    // stride 40 halves
  __shared__ alignas(16) _Float16 BtL[2][128 * 40];   // transposed [n][k]

  const int tid  = threadIdx.x;
  const int lane = tid & 31;
  const int wave = tid >> 5;
  const int ln   = lane & 15;
  const int g    = lane >> 4;
  const int m0   = blockIdx.x * 16;
  const int n0   = blockIdx.y * 128;

  // ---- hoisted A loader: 2 consecutive elements of one row --------------
  const int ae = tid * 2;
  const int ar = ae >> 5;
  const int ac = ae & 31;
  const float* arowp;
  {
    const int gr = m0 + ar;
    if (MODE == 0) {
      arowp = A + (size_t)gr * 256 + ac;
    } else {
      const int bb = gr / TOKPB;
      const int pp = gr - bb * TOKPB;
      const int y  = pp / HQ;
      const int x  = pp - y * HQ;
      arowp = A + (((size_t)bb * PHQ + y) * PHQ + x) * 256 + ac;
    }
  }
  const int aStOff = ar * 40 + ac;            // half-index in Ah row

  // ---- hoisted B loader: 8 (k-pair, n) items per thread ------------------
  const float* pB[8];
  int bStOff[8];
#pragma unroll
  for (int i = 0; i < 8; ++i) {
    const int e = tid + i * 256;              // 0..2047
    const int k = (e >> 7) * 2;               // even k in 0..30
    const int n = e & 127;
    pB[i]     = B + (size_t)k * N + n0 + n;
    bStOff[i] = n * 40 + k;
  }

  // ---- prologue: stage tile 0 --------------------------------------------
  {
    const float2 a2 = *(const float2*)(arowp);
    *(unsigned*)(&AhL[0][aStOff]) = pack2h(a2.x, a2.y);
#pragma unroll
    for (int i = 0; i < 8; ++i) {
      const float b0v = pB[i][0];
      const float b1v = pB[i][N];
      *(unsigned*)(&BtL[0][bStOff[i]]) = pack2h(b0v, b1v);
    }
  }

  v8f acc = {};
#pragma unroll
  for (int ks = 0; ks < 8; ++ks) {
    __syncthreads();
    const int cur = ks & 1;
    const int nxt = cur ^ 1;

    // issue next tile's global loads into registers (latency hidden by WMMA)
    float2 aN;
    float  bN0[8], bN1[8];
    if (ks < 7) {
      const int k1 = (ks + 1) * 32;
      aN = *(const float2*)(arowp + k1);
#pragma unroll
      for (int i = 0; i < 8; ++i) {
        bN0[i] = pB[i][(size_t)k1 * N];
        bN1[i] = pB[i][(size_t)k1 * N + N];
      }
    }

    // fragment loads + WMMA on current buffer
    FragU aF, bF;
    const _Float16* ap = &AhL[cur][ln * 40];
    aF.h[0] = *(const v8h*)(ap + g * 8);            // K = g*8 ..
    aF.h[1] = *(const v8h*)(ap + 16 + g * 8);       // K = 16+g*8 ..
    const _Float16* bp = &BtL[cur][(wave * 16 + ln) * 40];
    bF.h[0] = *(const v8h*)(bp + g * 8);            // K = g*8 ..
    bF.h[1] = *(const v8h*)(bp + 16 + g * 8);
    acc = wmma_f16(aF, bF, acc);

    // convert + store next tile into the other buffer
    if (ks < 7) {
      *(unsigned*)(&AhL[nxt][aStOff]) = pack2h(aN.x, aN.y);
#pragma unroll
      for (int i = 0; i < 8; ++i)
        *(unsigned*)(&BtL[nxt][bStOff[i]]) = pack2h(bN0[i], bN1[i]);
    }
  }

  const int col = n0 + wave * 16 + ln;
#pragma unroll
  for (int r = 0; r < 8; ++r) {
    const int grow = m0 + g * 8 + r;
    C[(size_t)grow * N + col] = acc[r];
  }
}

// ---------------------------------------------------------------------------
// Depthwise 3x3 conv (SAME) + LayerNorm(768) + q/v bias, writing into the
// zero-padded 192x192 buffer. One block (256 thr) per output pixel.
// ---------------------------------------------------------------------------
__global__ __launch_bounds__(256) void dwconv_ln_kernel(
    const float* __restrict__ qkv_raw, const float* __restrict__ dw_w,
    const float* __restrict__ ln_g, const float* __restrict__ ln_b,
    const float* __restrict__ q_bias, const float* __restrict__ v_bias,
    float* __restrict__ qkvp)
{
  const int tid = threadIdx.x;
  const int pix = blockIdx.x;                 // b*192*192 + y*192 + x
  const int b   = pix / (PHQ * PHQ);
  const int rem = pix - b * (PHQ * PHQ);
  const int y   = rem / PHQ;
  const int x   = rem - y * PHQ;
  float* out = qkvp + (size_t)pix * 768;

  if (y >= HQ || x >= HQ) {                   // zero pad region
#pragma unroll
    for (int i = 0; i < 3; ++i) out[tid + i * 256] = 0.0f;
    return;
  }

  float v[3];
#pragma unroll
  for (int i = 0; i < 3; ++i) {
    const int c = tid + i * 256;
    float s = 0.0f;
    for (int dy = -1; dy <= 1; ++dy) {
      const int yy = y + dy;
      if (yy < 0 || yy >= HQ) continue;
      for (int dx = -1; dx <= 1; ++dx) {
        const int xx = x + dx;
        if (xx < 0 || xx >= HQ) continue;
        s += qkv_raw[(((size_t)b * HQ + yy) * HQ + xx) * 768 + c] *
             dw_w[((dy + 1) * 3 + (dx + 1)) * 768 + c];
      }
    }
    v[i] = s;
  }

  __shared__ float s1[256];
  __shared__ float s2[256];
  s1[tid] = v[0] + v[1] + v[2];
  s2[tid] = v[0] * v[0] + v[1] * v[1] + v[2] * v[2];
  __syncthreads();
  for (int off = 128; off > 0; off >>= 1) {
    if (tid < off) { s1[tid] += s1[tid + off]; s2[tid] += s2[tid + off]; }
    __syncthreads();
  }
  const float mu  = s1[0] * (1.0f / 768.0f);
  const float var = s2[0] * (1.0f / 768.0f) - mu * mu;
  const float inv = rsqrtf(var + 1e-5f);

#pragma unroll
  for (int i = 0; i < 3; ++i) {
    const int c = tid + i * 256;
    float val = (v[i] - mu) * inv * ln_g[c] + ln_b[c];
    if (c < 256)       val += q_bias[c];
    else if (c >= 512) val += v_bias[c - 512];
    out[c] = val;
  }
}

// ---------------------------------------------------------------------------
// CPB MLP: table529[529][8] = 16*sigmoid(gelu(REL_TAB @ w0 + b0) @ w1).
// One block per table row; REL_TAB recomputed on device.
// ---------------------------------------------------------------------------
__device__ __forceinline__ float rel_coord(int t) {
  const float raw = (float)(t - 11) * (8.0f / 7.0f);
  const float mag = log1pf(fabsf(raw)) * (1.0f / 2.0794415416798357f); // /ln(8)
  return raw < 0.0f ? -mag : mag;
}

__global__ __launch_bounds__(256) void cpb_kernel(
    const float* __restrict__ w0, const float* __restrict__ b0,
    const float* __restrict__ w1, float* __restrict__ table)
{
  const int i   = blockIdx.x;      // 0..528
  const int tid = threadIdx.x;
  const float f0 = rel_coord(i / 23);
  const float f1 = rel_coord(i % 23);

  float p[8];
#pragma unroll
  for (int h = 0; h < 8; ++h) p[h] = 0.0f;

#pragma unroll
  for (int rep = 0; rep < 2; ++rep) {
    const int u = tid + rep * 256;
    const float hh = f0 * w0[u] + f1 * w0[512 + u] + b0[u];
    const float gl = 0.5f * hh * (1.0f + erff(hh * 0.70710678f));
#pragma unroll
    for (int h = 0; h < 8; ++h) p[h] += gl * w1[u * 8 + h];
  }

  __shared__ float red[256];
  for (int h = 0; h < 8; ++h) {
    red[tid] = p[h];
    __syncthreads();
    for (int off = 128; off > 0; off >>= 1) {
      if (tid < off) red[tid] += red[tid + off];
      __syncthreads();
    }
    if (tid == 0) table[i * 8 + h] = 16.0f / (1.0f + __expf(-red[0]));
    __syncthreads();
  }
}

// ---------------------------------------------------------------------------
// Halo window attention. One block (256 thr = 8 waves) per (b, window, head).
// Wave w: q-tile qi = w&3 (16 rows), k-half par = w>>2 (128 of 256 k-cols).
// QK^T via 8 WMMAs/wave, softmax in registers (shfl_xor over 16-lane halves
// + tiny LDS exchange between the 2 waves sharing a q-tile), P@V via 8
// WMMAs/wave. P buffer in LDS aliases the dead q/k staging region. The
// per-head CPB bias column is staged in LDS.
// ---------------------------------------------------------------------------
__global__ __launch_bounds__(256) void attn_kernel(
    const float* __restrict__ qkvp, const float* __restrict__ scale,
    const float* __restrict__ table, float* __restrict__ attn_out)
{
  __shared__ alignas(16) unsigned char smem[52352];
  _Float16* qh   = (_Float16*)smem;              // [64][32]  (phase 1)
  _Float16* kh   = (_Float16*)(smem + 4096);     // [256][32] (phase 1)
  _Float16* Pall = (_Float16*)smem;              // [64][256] (phase 2, alias)
  _Float16* vth  = (_Float16*)(smem + 32768);    // [32][256] V transposed
  float*    pmax = (float*)(smem + 49152);       // [2][64]
  float*    psum = pmax + 128;                   // [2][64]
  float*    tblh = (float*)(smem + 50176);       // [529] per-head CPB column

  const int tid  = threadIdx.x;
  const int lane = tid & 31;
  const int wave = tid >> 5;
  const int ln   = lane & 15;
  const int g    = lane >> 4;

  const int win  = blockIdx.x;
  const int head = blockIdx.y;
  const int b    = blockIdx.z;
  const int wy   = win / NWIN;
  const int wx   = win - wy * NWIN;

  const float* base = qkvp + (size_t)b * PHQ * PHQ * 768;
  const float  ls   = __expf(fminf(scale[head], 4.6051702f));  // log(100)

  // ---- stage per-head CPB table column ----------------------------------
  for (int i = tid; i < 529; i += 256) tblh[i] = table[i * 8 + head];

  // ---- stage q: l2-normalize + logit scale, f16 -------------------------
  {
    const int qr = tid >> 2, part = tid & 3;
    const int y = wy * 8 + (qr >> 3), x = wx * 8 + (qr & 7);
    const float* p = base + ((size_t)y * PHQ + x) * 768 + head * 32 + part * 8;
    const float4 a4 = *(const float4*)p;
    const float4 b4 = *(const float4*)(p + 4);
    float ss = a4.x*a4.x + a4.y*a4.y + a4.z*a4.z + a4.w*a4.w +
               b4.x*b4.x + b4.y*b4.y + b4.z*b4.z + b4.w*b4.w;
    ss += __shfl_xor(ss, 1, 32);
    ss += __shfl_xor(ss, 2, 32);
    const float sc = ls * rsqrtf(fmaxf(ss, 1.55e-5f));
    v8h o;
    o[0]=(_Float16)(a4.x*sc); o[1]=(_Float16)(a4.y*sc);
    o[2]=(_Float16)(a4.z*sc); o[3]=(_Float16)(a4.w*sc);
    o[4]=(_Float16)(b4.x*sc); o[5]=(_Float16)(b4.y*sc);
    o[6]=(_Float16)(b4.z*sc); o[7]=(_Float16)(b4.w*sc);
    *(v8h*)(qh + qr * 32 + part * 8) = o;
  }
  // ---- stage k (l2-normalized) and V (transposed), f16 ------------------
  {
    const int kr = tid;                       // halo position 0..255
    const int gy = wy * 8 - 4 + (kr >> 4);
    const int gx = wx * 8 - 4 + (kr & 15);
    const bool inb = (gy >= 0 && gy < PHQ && gx >= 0 && gx < PHQ);
    float kv[32], vv[32];
    if (inb) {
      const float* pk = base + ((size_t)gy * PHQ + gx) * 768 + 256 + head * 32;
      const float* pv = base + ((size_t)gy * PHQ + gx) * 768 + 512 + head * 32;
#pragma unroll
      for (int j = 0; j < 8; ++j) {
        const float4 t = *(const float4*)(pk + j * 4);
        kv[j*4+0]=t.x; kv[j*4+1]=t.y; kv[j*4+2]=t.z; kv[j*4+3]=t.w;
        const float4 u = *(const float4*)(pv + j * 4);
        vv[j*4+0]=u.x; vv[j*4+1]=u.y; vv[j*4+2]=u.z; vv[j*4+3]=u.w;
      }
    } else {
#pragma unroll
      for (int j = 0; j < 32; ++j) { kv[j] = 0.0f; vv[j] = 0.0f; }
    }
    float ss = 0.0f;
#pragma unroll
    for (int j = 0; j < 32; ++j) ss += kv[j] * kv[j];
    const float sc = rsqrtf(fmaxf(ss, 1.55e-5f));
#pragma unroll
    for (int j = 0; j < 4; ++j) {
      v8h o;
#pragma unroll
      for (int e = 0; e < 8; ++e) o[e] = (_Float16)(kv[j * 8 + e] * sc);
      *(v8h*)(kh + kr * 32 + j * 8) = o;
    }
#pragma unroll
    for (int d = 0; d < 32; ++d) vth[d * 256 + kr] = (_Float16)vv[d];
  }
  __syncthreads();

  // ---- S = q_n * scale @ k_n^T : 8 WMMAs per wave -----------------------
  const int qi  = wave & 3;
  const int par = wave >> 2;                  // 0 or 1 (k columns half)

  FragU aF;
  {
    const _Float16* ap = qh + (qi * 16 + ln) * 32;
    aF.h[0] = *(const v8h*)(ap + g * 8);
    aF.h[1] = *(const v8h*)(ap + 16 + g * 8);
  }
  v8f accs[8];
#pragma unroll
  for (int j = 0; j < 8; ++j) {
    const int kj = par * 8 + j;
    FragU bF;
    const _Float16* bp = kh + (kj * 16 + ln) * 32;
    bF.h[0] = *(const v8h*)(bp + g * 8);
    bF.h[1] = *(const v8h*)(bp + 16 + g * 8);
    v8f z = {};
    accs[j] = wmma_f16(aF, bF, z);
  }

  // ---- add CPB bias (from LDS) + halo mask in-register ------------------
#pragma unroll
  for (int j = 0; j < 8; ++j) {
    const int kcol = par * 128 + j * 16 + ln;
    const int ky = kcol >> 4, kx = kcol & 15;
    const int gy = wy * 8 - 4 + ky, gx = wx * 8 - 4 + kx;
    const float mb = (gy >= 0 && gy < HQ && gx >= 0 && gx < HQ) ? 0.0f : -100.0f;
#pragma unroll
    for (int r = 0; r < 8; ++r) {
      const int qrow = qi * 16 + g * 8 + r;
      const int idx  = ((qrow >> 3) - ky + 15) * 23 + ((qrow & 7) - kx + 15);
      accs[j][r] += tblh[idx] + mb;
    }
  }

  // ---- softmax over 256 k-cols ------------------------------------------
  float fm[8];
#pragma unroll
  for (int r = 0; r < 8; ++r) {
    float m = accs[0][r];
#pragma unroll
    for (int j = 1; j < 8; ++j) m = fmaxf(m, accs[j][r]);
    m = fmaxf(m, __shfl_xor(m, 1, 32));
    m = fmaxf(m, __shfl_xor(m, 2, 32));
    m = fmaxf(m, __shfl_xor(m, 4, 32));
    m = fmaxf(m, __shfl_xor(m, 8, 32));
    fm[r] = m;
  }
  if (ln == 0) {
#pragma unroll
    for (int r = 0; r < 8; ++r) pmax[par * 64 + qi * 16 + g * 8 + r] = fm[r];
  }
  __syncthreads();
#pragma unroll
  for (int r = 0; r < 8; ++r) {
    const int row = qi * 16 + g * 8 + r;
    fm[r] = fmaxf(pmax[row], pmax[64 + row]);
  }
  float ps[8];
#pragma unroll
  for (int r = 0; r < 8; ++r) {
    float s = 0.0f;
#pragma unroll
    for (int j = 0; j < 8; ++j) {
      const float e = __expf(accs[j][r] - fm[r]);
      accs[j][r] = e;
      s += e;
    }
    s += __shfl_xor(s, 1, 32);
    s += __shfl_xor(s, 2, 32);
    s += __shfl_xor(s, 4, 32);
    s += __shfl_xor(s, 8, 32);
    ps[r] = s;
  }
  if (ln == 0) {
#pragma unroll
    for (int r = 0; r < 8; ++r) psum[par * 64 + qi * 16 + g * 8 + r] = ps[r];
  }
  __syncthreads();
#pragma unroll
  for (int r = 0; r < 8; ++r) {
    const int row = qi * 16 + g * 8 + r;
    const float inv = 1.0f / (psum[row] + psum[64 + row]);
#pragma unroll
    for (int j = 0; j < 8; ++j) accs[j][r] *= inv;
  }

  // ---- write P (f16) into the aliased LDS buffer ------------------------
#pragma unroll
  for (int j = 0; j < 8; ++j) {
#pragma unroll
    for (int r = 0; r < 8; ++r) {
      const int row = qi * 16 + g * 8 + r;
      Pall[row * 256 + par * 128 + j * 16 + ln] = (_Float16)accs[j][r];
    }
  }
  __syncthreads();

  // ---- O = P @ V : 8 WMMAs per wave -------------------------------------
  const int oqi = wave & 3;
  const int oni = (wave >> 2) & 1;
  v8f o = {};
  const _Float16* prow = Pall + (oqi * 16 + ln) * 256;
  const _Float16* vrow = vth + (oni * 16 + ln) * 256;
#pragma unroll
  for (int ksb = 0; ksb < 8; ++ksb) {
    FragU a2, b2;
    a2.h[0] = *(const v8h*)(prow + ksb * 32 + g * 8);
    a2.h[1] = *(const v8h*)(prow + ksb * 32 + 16 + g * 8);
    b2.h[0] = *(const v8h*)(vrow + ksb * 32 + g * 8);
    b2.h[1] = *(const v8h*)(vrow + ksb * 32 + 16 + g * 8);
    o = wmma_f16(a2, b2, o);
  }

#pragma unroll
  for (int r = 0; r < 8; ++r) {
    const int qrow = oqi * 16 + g * 8 + r;
    const int y = wy * 8 + (qrow >> 3);
    const int x = wx * 8 + (qrow & 7);
    const int d = oni * 16 + ln;
    attn_out[(((size_t)b * PHQ + y) * PHQ + x) * 256 + head * 32 + d] = o[r];
  }
}

// ---------------------------------------------------------------------------
// Launch
// ---------------------------------------------------------------------------
extern "C" void kernel_launch(void* const* d_in, const int* in_sizes, int n_in,
                              void* d_out, int out_size, void* d_ws, size_t ws_size,
                              hipStream_t stream) {
  (void)in_sizes; (void)n_in; (void)out_size; (void)ws_size;
  const float* x      = (const float*)d_in[0];
  const float* qkv_w  = (const float*)d_in[1];
  const float* dw_w   = (const float*)d_in[2];
  const float* ln_g   = (const float*)d_in[3];
  const float* ln_b   = (const float*)d_in[4];
  const float* q_bias = (const float*)d_in[5];
  const float* v_bias = (const float*)d_in[6];
  const float* scale  = (const float*)d_in[7];
  const float* cpb_w0 = (const float*)d_in[8];
  const float* cpb_b0 = (const float*)d_in[9];
  const float* cpb_w1 = (const float*)d_in[10];
  const float* proj_w = (const float*)d_in[11];

  float* ws       = (float*)d_ws;
  float* qkv_raw  = ws;                              // 70688*768
  float* qkvp     = qkv_raw + (size_t)TOK * 768;     // 2*192*192*768
  float* attn_o   = qkvp + (size_t)BQ * PHQ * PHQ * 768;   // 2*192*192*256
  float* table    = attn_o + (size_t)BQ * PHQ * PHQ * 256; // 529*8

  // 1) QKV GEMM: [70688 x 256] @ [256 x 768]
  gemm16x128_kernel<768, 0><<<dim3(TOK / 16, 6), 256, 0, stream>>>(x, qkv_w, qkv_raw);
  // 2) depthwise conv + LN + bias, into zero-padded 192x192 buffer
  dwconv_ln_kernel<<<dim3(BQ * PHQ * PHQ), 256, 0, stream>>>(
      qkv_raw, dw_w, ln_g, ln_b, q_bias, v_bias, qkvp);
  // 3) CPB bias table
  cpb_kernel<<<dim3(529), 256, 0, stream>>>(cpb_w0, cpb_b0, cpb_w1, table);
  // 4) halo window attention (one block per b,window,head)
  attn_kernel<<<dim3(NWIN * NWIN, 8, BQ), 256, 0, stream>>>(qkvp, scale, table, attn_o);
  // 5) output projection with crop gather: [70688 x 256] @ [256 x 256]
  gemm16x128_kernel<256, 1><<<dim3(TOK / 16, 2), 256, 0, stream>>>(attn_o, proj_w, (float*)d_out);
}
